// NodeModel_50371376447827
// MI455X (gfx1250) — matrix operand order repaired and checked by reading, not compiled, not run on hardware
//
#include <hip/hip_runtime.h>

// ---------------------------------------------------------------------------
// Problem constants (from the reference): N nodes, E edges, D feat, H hidden,
// O out.  N = 6250 * 16 exactly, so M-tiles of 16 need no tail handling.
// ---------------------------------------------------------------------------
#define N_NODES 100000
#define N_EDGES 1600000
#define D_FEAT  64
#define H_DIM   256
#define O_DIM   64
#define TILES   (N_NODES / 16)        // 6250

#define W1_FRAG_HALFS (64 * 32 * 16)  // 32768 bf16 = 64 KB
#define W2_FRAG_HALFS (32 * 32 * 16)  // 16384 bf16 = 32 KB
#define WPOOL_BYTES   ((W1_FRAG_HALFS + W2_FRAG_HALFS) * 2)   // 98304 B

typedef __attribute__((ext_vector_type(16))) __bf16 v16bf;
typedef __attribute__((ext_vector_type(8)))  float  v8f;

// ---------------------------------------------------------------------------
// Kernel 1: zero the segment-sum accumulator (float4 stores, 1.6M threads).
// ---------------------------------------------------------------------------
__global__ void zero_agg(float4* __restrict__ agg4) {
  int t = blockIdx.x * blockDim.x + threadIdx.x;
  if (t < N_NODES * D_FEAT / 4) {
    float4 z = {0.f, 0.f, 0.f, 0.f};
    agg4[t] = z;
  }
}

// ---------------------------------------------------------------------------
// Kernel 2: repack W1 [128,256] and W2 [256,64] (f32 row-major, k-major) into
// a single contiguous bf16 WMMA B-fragment pool.  ISA 16-bit B(32x16) layout:
//   lanes 0-15 : N = lane,    K = 0..7, 16..23
//   lanes 16-31: N = lane-16, K = 8..15, 24..31
// Each lane's 16 bf16 values are contiguous (32 B) -> one 2xb128 fetch.
// ---------------------------------------------------------------------------
__global__ void prep_weights(const float* __restrict__ W1,
                             const float* __restrict__ W2,
                             __bf16* __restrict__ wpool) {
  int t = blockIdx.x * blockDim.x + threadIdx.x;
  if (t < 64 * 32) {                      // W1: 16 n-tiles x 4 k-tiles
    int f = t >> 5, lane = t & 31;
    int nt = f >> 2, kt = f & 3;
    int col = nt * 16 + (lane & 15);
    int kb  = kt * 32 + ((lane & 16) ? 8 : 0);
    __bf16* dst = wpool + t * 16;
#pragma unroll
    for (int i = 0; i < 16; ++i)
      dst[i] = (__bf16)W1[(kb + i + (i >= 8 ? 8 : 0)) * H_DIM + col];
  } else if (t < 64 * 32 + 32 * 32) {     // W2: 4 n-tiles x 8 k-tiles
    int u = t - 64 * 32;
    int f = u >> 5, lane = u & 31;
    int ot = f >> 3, j = f & 7;
    int col = ot * 16 + (lane & 15);
    int kb  = j * 32 + ((lane & 16) ? 8 : 0);
    __bf16* dst = wpool + W1_FRAG_HALFS + u * 16;
#pragma unroll
    for (int i = 0; i < 16; ++i)
      dst[i] = (__bf16)W2[(kb + i + (i >= 8 ? 8 : 0)) * O_DIM + col];
  }
}

// ---------------------------------------------------------------------------
// Kernel 3: scatter-add edge features into agg.  4 threads per edge, each
// handling 16 consecutive floats via float4 loads + f32 global atomics.
// This phase is the HBM-bandwidth roofline of the whole problem (~410 MB).
// ---------------------------------------------------------------------------
__global__ void scatter_edges(const float* __restrict__ edge_attr,
                              const int* __restrict__ row,
                              float* __restrict__ agg) {
  int gid = blockIdx.x * blockDim.x + threadIdx.x;
  int e = gid >> 2;
  int c = (gid & 3) * 16;
  if (e >= N_EDGES) return;
  int r = row[e];                                   // edge_index[0][e]
  const float4* s = (const float4*)(edge_attr + (size_t)e * D_FEAT + c);
  float* d = agg + (size_t)r * D_FEAT + c;
#pragma unroll
  for (int i = 0; i < 4; ++i) {
    float4 v = s[i];
    atomicAdd(d + i * 4 + 0, v.x);
    atomicAdd(d + i * 4 + 1, v.y);
    atomicAdd(d + i * 4 + 2, v.z);
    atomicAdd(d + i * 4 + 3, v.w);
  }
}

// ---------------------------------------------------------------------------
// Kernel 4: fused MLP.  One wave32 per 16-node tile; 8 waves / 256-thread
// block.  The 96 KB weight-fragment pool is staged into LDS ONCE per block
// with the CDNA5 async copy engine (global_load_async_to_lds_b128, ASYNCcnt),
// then all WMMA B operands come from ds_load_b128.  GEMM1 (K=128) produces
// 32-wide hidden slabs (2 n-tiles x 4 k-steps), bias+ReLU fused, staged bf16
// in a wave-private LDS buffer (C/D layout -> A layout cross-lane transpose;
// row stride 40 bf16 = 20 dwords == 4 mod 64 banks -> conflict-free b128
// reads), then immediately consumed by GEMM2 (K=256) into 4 output tiles.
// ---------------------------------------------------------------------------
__global__ __launch_bounds__(256) void fused_mlp(
    const float* __restrict__ x, const float* __restrict__ agg,
    const __bf16* __restrict__ wpool,
    const float* __restrict__ b1, const float* __restrict__ b2,
    float* __restrict__ out) {
  __shared__ __attribute__((aligned(32))) __bf16 wbuf[W1_FRAG_HALFS + W2_FRAG_HALFS];
  __shared__ __bf16 hslab[8][16 * 40];     // 10,240 B

  int wave = threadIdx.x >> 5;
  int lane = threadIdx.x & 31;
  int tile = blockIdx.x * 8 + wave;

  // ---- Stage the whole weight pool into LDS with async b128 copies.
  // 98304 B / 16 B = 6144 transfers; 256 threads x 24 iterations.
  {
    uint32_t lbase = (uint32_t)(uintptr_t)(&wbuf[0]);   // low 32 bits = LDS offset
#pragma unroll 4
    for (int it = 0; it < 24; ++it) {
      uint32_t off = (uint32_t)threadIdx.x * 16u + (uint32_t)it * 4096u;
      uint32_t lds_addr = lbase + off;
      asm volatile("global_load_async_to_lds_b128 %0, %1, %2"
                   :
                   : "v"(lds_addr), "v"(off), "s"(wpool)
                   : "memory");
    }
    asm volatile("s_wait_asynccnt 0" ::: "memory");
  }
  __syncthreads();                         // all waves see staged weights

  if (tile < TILES) {                      // wave-uniform: EXEC stays all-1s
    int m0 = tile * 16;
    int r    = lane & 15;                  // row (A) / column (B,C,D) id
    int hsel = lane >> 4;                  // half-wave select

    // ---- A fragments for GEMM1: [x | agg] rows m0..m0+15, K = 0..127, bf16.
    // ISA 16-bit A(16x32) layout: lane<16 row=lane holds K kb..kb+7 and
    // kb+16..kb+23 with kb=0; lanes>=16 same row, kb=8.
    v16bf a1[4];
#pragma unroll
    for (int kt = 0; kt < 4; ++kt) {
      const float* p = (kt < 2)
          ? (x   + (size_t)(m0 + r) * D_FEAT + kt * 32)
          : (agg + (size_t)(m0 + r) * D_FEAT + (kt - 2) * 32);
      int kb = hsel * 8;
#pragma unroll
      for (int i = 0; i < 8; ++i) {
        a1[kt][i]     = (__bf16)p[kb + i];
        a1[kt][8 + i] = (__bf16)p[kb + 16 + i];
      }
    }

    v8f acc2[4] = {};                      // out tile accumulators (O = 4x16)

    for (int j = 0; j < 8; ++j) {          // 8 hidden slabs of 32 columns
      // ---- GEMM1: hidden cols [j*32, j*32+32), bias + ReLU, bf16 -> slab.
#pragma unroll
      for (int s = 0; s < 2; ++s) {
        int nt = 2 * j + s;
        v8f c = {};
#pragma unroll
        for (int kt = 0; kt < 4; ++kt) {
          v16bf b = *(const v16bf*)(wbuf + ((nt * 4 + kt) * 32 + lane) * 16);
          c = __builtin_amdgcn_wmma_f32_16x16x32_bf16(
                  false, a1[kt], false, b, (short)0, c, false, false);
        }
        float bias = b1[nt * 16 + r];      // D column = r for this lane
        int col = s * 16 + r;
#pragma unroll
        for (int rr = 0; rr < 8; ++rr) {   // D row = rr + 8*hsel
          float v = c[rr] + bias;
          v = v > 0.f ? v : 0.f;
          hslab[wave][(hsel * 8 + rr) * 40 + col] = (__bf16)v;
        }
      }

      // ---- Reload slab in A layout (wave-private LDS: DS ops in-order per
      //      wave; compiler inserts s_wait_dscnt for the RAW, no barrier).
      v16bf a2;
      {
        const __bf16* hp = &hslab[wave][r * 40 + hsel * 8];
#pragma unroll
        for (int i = 0; i < 8; ++i) {
          a2[i]     = hp[i];
          a2[8 + i] = hp[16 + i];
        }
      }

      // ---- GEMM2: accumulate this K=32 slab into the 4 output tiles.
#pragma unroll
      for (int ot = 0; ot < 4; ++ot) {
        v16bf b = *(const v16bf*)(wbuf + W1_FRAG_HALFS +
                                  ((ot * 8 + j) * 32 + lane) * 16);
        acc2[ot] = __builtin_amdgcn_wmma_f32_16x16x32_bf16(
                       false, a2, false, b, (short)0, acc2[ot], false, false);
      }
    }

    // ---- Epilogue: + b2, coalesced f32 stores (lanes 0-15 cover 64 B rows).
#pragma unroll
    for (int ot = 0; ot < 4; ++ot) {
      float bias = b2[ot * 16 + r];
      int col = ot * 16 + r;
#pragma unroll
      for (int rr = 0; rr < 8; ++rr) {
        out[(size_t)(m0 + hsel * 8 + rr) * O_DIM + col] = acc2[ot][rr] + bias;
      }
    }
  }
}

// ---------------------------------------------------------------------------
// Host launcher.  Inputs (setup_inputs order):
//   0:x [N,64] f32   1:edge_index [2,E] int   2:edge_attr [E,64] f32
//   3:u (unused)     4:batch (unused)
//   5:W1 [128,256]   6:b1 [256]   7:W2 [256,64]   8:b2 [64]
// Workspace: agg (25.6 MB f32) | weight-fragment pool (96 KB bf16).
// ---------------------------------------------------------------------------
extern "C" void kernel_launch(void* const* d_in, const int* in_sizes, int n_in,
                              void* d_out, int out_size, void* d_ws, size_t ws_size,
                              hipStream_t stream) {
  const float* x   = (const float*)d_in[0];
  const int*  eidx = (const int*) d_in[1];
  const float* ea  = (const float*)d_in[2];
  const float* W1  = (const float*)d_in[5];
  const float* b1  = (const float*)d_in[6];
  const float* W2  = (const float*)d_in[7];
  const float* b2  = (const float*)d_in[8];
  float* out = (float*)d_out;

  float* agg = (float*)d_ws;
  __bf16* wpool = (__bf16*)((char*)d_ws + (size_t)N_NODES * D_FEAT * sizeof(float));

  zero_agg<<<(N_NODES * D_FEAT / 4 + 255) / 256, 256, 0, stream>>>((float4*)agg);
  prep_weights<<<12, 256, 0, stream>>>(W1, W2, wpool);
  scatter_edges<<<(N_EDGES * 4) / 256, 256, 0, stream>>>(ea, eidx, agg);
  fused_mlp<<<(TILES + 7) / 8, 256, 0, stream>>>(x, agg, wpool, b1, b2, out);
}
